// Model_20547123544443
// MI455X (gfx1250) — compile-verified, matrix operand order checked by reading, not run on hardware
//
#include <hip/hip_runtime.h>
#include <hip/hip_bf16.h>

typedef __bf16 bf16;
typedef __attribute__((ext_vector_type(16))) __bf16 bf16x16;
typedef __attribute__((ext_vector_type(8)))  __bf16 bf16x8;
typedef __attribute__((ext_vector_type(4)))  __bf16 bf16x4;
typedef __attribute__((ext_vector_type(8)))  float  f32x8;
typedef __attribute__((ext_vector_type(4)))  float  f32x4;

namespace {
constexpr int kB  = 4;
constexpr int kT  = 2048;
constexpr int kC  = 768;
constexpr int kC3 = 2304;
constexpr int kH  = 12;
constexpr int kDh = 64;
constexpr float kScale = 0.125f;   // 1/sqrt(Dh)
}

__device__ __forceinline__ f32x8 wmma_bf16(bf16x16 a, bf16x16 b, f32x8 c) {
  // D = A(16x32 bf16) * B(32x16 bf16) + C(16x16 f32)
  return __builtin_amdgcn_wmma_f32_16x16x32_bf16(
      /*neg_a=*/false, a, /*neg_b=*/false, b,
      /*c_mod=*/(short)0, c, /*reuse_a=*/false, /*reuse_b=*/false);
}

__device__ __forceinline__ f32x8 zero8() {
  f32x8 z = {0.f, 0.f, 0.f, 0.f, 0.f, 0.f, 0.f, 0.f};
  return z;
}

union Frag { bf16x16 v; bf16x8 h[2]; };

// Per-lane fragment chunk pattern shared by 16-bit A(16x32) and B(32x16):
// elements 0..7  = K  8*hl .. 8*hl+7
// elements 8..15 = K 16+8*hl .. 16+8*hl+7
__device__ __forceinline__ bf16x16 load_frag(const bf16* row, int hl) {
  Frag f;
  f.h[0] = *(const bf16x8*)(row + 8 * hl);
  f.h[1] = *(const bf16x8*)(row + 16 + 8 * hl);
  return f.v;
}

// ---------------------------------------------------------------------------
// Kernel 1: qkv = x @ W + b  (f32 in, bf16 out; q scaled; v stored transposed)
// Block tile 128x128, K-step 32, 8 waves: wm in {0,1} x 64 rows, wn in {0..3} x 32 cols
// ---------------------------------------------------------------------------
__global__ __launch_bounds__(256) void qkv_gemm_kernel(
    const float* __restrict__ x, const float* __restrict__ W,
    const float* __restrict__ bias,
    bf16* __restrict__ qws, bf16* __restrict__ kws, bf16* __restrict__ vws)
{
  __shared__ bf16 lds_x[128 * 32];   // [m][k]
  __shared__ bf16 lds_w[128 * 32];   // transposed: [n][k]

  const int tid  = threadIdx.x;
  const int lane = tid & 31;
  const int wave = tid >> 5;
  const int wm   = wave & 1;         // 0..1 : 64-row slab
  const int wn   = wave >> 1;        // 0..3 : 32-col slab
  const int bt0  = blockIdx.x * 128; // global token block
  const int n0   = blockIdx.y * 128; // output-column block

  f32x8 acc[4][2];
#pragma unroll
  for (int mt = 0; mt < 4; ++mt)
#pragma unroll
    for (int nt = 0; nt < 2; ++nt) acc[mt][nt] = zero8();

  const int ml = lane & 15;
  const int hl = lane >> 4;

  for (int kk = 0; kk < kC; kk += 32) {
    // ---- stage x tile (128x32) into LDS as bf16, row-major [m][k]
    {
      const int r = tid >> 3;            // 0..31
      const int c = (tid & 7) * 4;       // 0..28
#pragma unroll
      for (int rr = 0; rr < 4; ++rr) {
        const int row = r + rr * 32;
        const f32x4 xv = *(const f32x4*)(x + (size_t)(bt0 + row) * kC + kk + c);
        bf16x4 b4;
#pragma unroll
        for (int j = 0; j < 4; ++j) b4[j] = (bf16)xv[j];
        *(bf16x4*)(lds_x + row * 32 + c) = b4;
      }
    }
    // ---- stage W tile (32x128) into LDS transposed as bf16 [n][k]
    {
      const int k  = tid >> 3;           // 0..31
      const int nb = (tid & 7) * 16;     // 0..112
#pragma unroll
      for (int i = 0; i < 4; ++i) {
        const f32x4 wv = *(const f32x4*)(W + (size_t)(kk + k) * kC3 + n0 + nb + i * 4);
#pragma unroll
        for (int j = 0; j < 4; ++j)
          lds_w[(nb + i * 4 + j) * 32 + k] = (bf16)wv[j];
      }
    }
    __syncthreads();

    bf16x16 afr[4], bfr[2];
#pragma unroll
    for (int mt = 0; mt < 4; ++mt)
      afr[mt] = load_frag(lds_x + (wm * 64 + mt * 16 + ml) * 32, hl);
#pragma unroll
    for (int nt = 0; nt < 2; ++nt)
      bfr[nt] = load_frag(lds_w + (wn * 32 + nt * 16 + ml) * 32, hl);
#pragma unroll
    for (int mt = 0; mt < 4; ++mt)
#pragma unroll
      for (int nt = 0; nt < 2; ++nt)
        acc[mt][nt] = wmma_bf16(afr[mt], bfr[nt], acc[mt][nt]);
    __syncthreads();
  }

  // ---- epilogue: bias, q-scale, scatter to q/k [B,H,T,Dh] and vT [B,H,Dh,T]
  const int b = bt0 >> 11;             // bt0 / T
#pragma unroll
  for (int nt = 0; nt < 2; ++nt) {
    const int ng  = n0 + wn * 32 + nt * 16 + ml;   // column in [0, 3C)
    const float bia = bias[ng];
    const int sec = ng / kC;                        // 0=q 1=k 2=v
    const int cg  = ng - sec * kC;
    const int h   = cg >> 6;
    const int d   = cg & 63;
#pragma unroll
    for (int mt = 0; mt < 4; ++mt) {
      const int t0 = (bt0 & (kT - 1)) + wm * 64 + mt * 16 + 8 * hl; // rows M = r + 8*hl
      if (sec == 2) {
        bf16x8 v8;
#pragma unroll
        for (int r = 0; r < 8; ++r) v8[r] = (bf16)(acc[mt][nt][r] + bia);
        *(bf16x8*)(vws + ((size_t)(b * kH + h) * kDh + d) * kT + t0) = v8;
      } else {
        bf16* dst = (sec == 0) ? qws : kws;
        const float scl = (sec == 0) ? kScale : 1.0f;
#pragma unroll
        for (int r = 0; r < 8; ++r)
          dst[((size_t)(b * kH + h) * kT + (t0 + r)) * kDh + d] =
              (bf16)((acc[mt][nt][r] + bia) * scl);
      }
    }
  }
}

// ---------------------------------------------------------------------------
// Kernel 2: causal ReLU attention, register-only P-transpose.
// One block per (b, h, 256 q-rows); each wave owns 32 q-rows (2 q-tiles).
// Scores are computed TRANSPOSED (S^T = K_tile x Q^T): the D-layout of two
// stacked 16-k S^T tiles is exactly the A-fragment layout of P(16q x 32k),
// so mask+ReLU+bf16-convert happens entirely in registers (no LDS).
// The resident q fragments serve directly as the B operand of S^T.
// ---------------------------------------------------------------------------
__global__ __launch_bounds__(256) void attn_kernel(
    const bf16* __restrict__ qws, const bf16* __restrict__ kws,
    const bf16* __restrict__ vws, float* __restrict__ out)
{
  const int tid  = threadIdx.x;
  const int lane = tid & 31;
  const int wave = tid >> 5;

  int idx = blockIdx.x;
  const int qb = idx & 7;   idx >>= 3;   // T/256 = 8
  const int h  = idx % kH;
  const int b  = idx / kH;

  const int qt0 = qb * 256 + wave * 32;  // this wave's 32 q-rows
  const int ml  = lane & 15;
  const int hl  = lane >> 4;
  const size_t bh = (size_t)(b * kH + h);

  // q fragments for 2 q-tiles x 2 K-windows; used as B operands of S^T
  bf16x16 aq[2][2];
#pragma unroll
  for (int qt = 0; qt < 2; ++qt) {
    const bf16* qrow = qws + (bh * kT + qt0 + qt * 16 + ml) * kDh;
    aq[qt][0] = load_frag(qrow, hl);
    aq[qt][1] = load_frag(qrow + 32, hl);
  }

  f32x8 yacc[2][4];
#pragma unroll
  for (int qt = 0; qt < 2; ++qt)
#pragma unroll
    for (int nd = 0; nd < 4; ++nd) yacc[qt][nd] = zero8();

  const int qmax = qt0 + 31;
  for (int kt0 = 0; kt0 <= qmax; kt0 += 32) {
    if (kt0 + 32 <= qmax) {
      // speculative prefetch of next k/v tiles (global_prefetch_b8)
      __builtin_prefetch(kws + (bh * kT + kt0 + 32 + ml) * kDh, 0, 1);
      __builtin_prefetch(kws + (bh * kT + kt0 + 48 + ml) * kDh, 0, 1);
      __builtin_prefetch(vws + (bh * kDh + ml * 4) * kT + kt0 + 32, 0, 1);
    }
    // k fragments: A operands of S^T (rows = k), shared by both q-tiles
    bf16x16 kf[2][2];
#pragma unroll
    for (int nt = 0; nt < 2; ++nt) {
      const bf16* krow = kws + (bh * kT + kt0 + nt * 16 + ml) * kDh;
      kf[nt][0] = load_frag(krow, hl);
      kf[nt][1] = load_frag(krow + 32, hl);
    }
    // v fragments: B operands of P*V, shared by both q-tiles
    bf16x16 vf[4];
#pragma unroll
    for (int nd = 0; nd < 4; ++nd)
      vf[nd] = load_frag(vws + (bh * kDh + nd * 16 + ml) * kT + kt0, hl);

#pragma unroll
    for (int qt = 0; qt < 2; ++qt) {
      // S^T tiles: st[j] covers k in [kt0+16j, kt0+16j+16), N = q rows
      f32x8 st0 = zero8(), st1 = zero8();
      st0 = wmma_bf16(kf[0][0], aq[qt][0], st0);
      st0 = wmma_bf16(kf[0][1], aq[qt][1], st0);
      st1 = wmma_bf16(kf[1][0], aq[qt][0], st1);
      st1 = wmma_bf16(kf[1][1], aq[qt][1], st1);

      // mask + ReLU + pack: S^T D-layout -> P A-fragment, all in registers.
      // lane: q = qt0+qt*16+ml (N), k = kt0 + r + 8*hl (M of tile j)
      const int qg = qt0 + qt * 16 + ml;
      Frag ap;
#pragma unroll
      for (int r = 0; r < 8; ++r) {
        const int kg0 = kt0 + r + 8 * hl;
        const int kg1 = kg0 + 16;
        const float s0 = (kg0 <= qg) ? fmaxf(st0[r], 0.f) : 0.f;
        const float s1 = (kg1 <= qg) ? fmaxf(st1[r], 0.f) : 0.f;
        ap.v[r]     = (bf16)s0;
        ap.v[r + 8] = (bf16)s1;
      }

#pragma unroll
      for (int nd = 0; nd < 4; ++nd)
        yacc[qt][nd] = wmma_bf16(ap.v, vf[nd], yacc[qt][nd]);
    }
  }

  // ---- write y [B,T,C] f32 (lane: d = nd*16+ml, rows t = base + r + 8*hl)
  float* obase = out + (size_t)b * kT * kC + (size_t)h * kDh;
#pragma unroll
  for (int qt = 0; qt < 2; ++qt) {
#pragma unroll
    for (int nd = 0; nd < 4; ++nd) {
      const int d = nd * 16 + ml;
#pragma unroll
      for (int r = 0; r < 8; ++r) {
        const int t = qt0 + qt * 16 + r + 8 * hl;
        obase[(size_t)t * kC + d] = yacc[qt][nd][r];
      }
    }
  }
}

// ---------------------------------------------------------------------------
extern "C" void kernel_launch(void* const* d_in, const int* in_sizes, int n_in,
                              void* d_out, int out_size, void* d_ws, size_t ws_size,
                              hipStream_t stream) {
  const float* x    = (const float*)d_in[0];   // [B,T,C]
  const float* W    = (const float*)d_in[1];   // [C,3C]
  const float* bias = (const float*)d_in[2];   // [3C]
  float* out = (float*)d_out;                  // [B,T,C]

  // workspace: q,k as [B,H,T,Dh] bf16, v transposed [B,H,Dh,T] bf16 (~36 MB)
  const size_t per = (size_t)kB * kH * kT * kDh;
  bf16* qws = (bf16*)d_ws;
  bf16* kws = qws + per;
  bf16* vws = kws + per;

  dim3 g1((kB * kT) / 128, kC3 / 128);         // 64 x 18 blocks
  qkv_gemm_kernel<<<g1, dim3(256), 0, stream>>>(x, W, bias, qws, kws, vws);

  dim3 g2(kB * kH * (kT / 256));               // 384 blocks
  attn_kernel<<<g2, dim3(256), 0, stream>>>(qws, kws, vws, out);
}
